// EfficientAttention_88665304858756
// MI455X (gfx1250) — matrix operand order repaired and checked by reading, not compiled, not run on hardware
//
#include <hip/hip_runtime.h>
#include <hip/hip_bf16.h>

typedef _Float16 f16;
typedef __attribute__((ext_vector_type(16))) _Float16 v16h;
typedef __attribute__((ext_vector_type(8)))  float    v8f;

#define WMMA_F32_F16(a, b, c) \
    __builtin_amdgcn_wmma_f32_16x16x32_f16(false, (a), false, (b), (short)0, (c), false, false)

// Problem constants
static constexpr int Bsz    = 2;
static constexpr int Sseq   = 2048;
static constexpr int HID    = 1024;
static constexpr int NH     = 16;
static constexpr int HD     = 64;
static constexpr float SCALE = 125.0f;  // TEMP_K / sqrt(HEAD_DIM) = 1000/8

// ---------------------------------------------------------------------------
// Kernel 1: QKV projection.  X[4096,1024] @ W[1024,3072] + b.
// Q,K stored f16 [B,NH,S,HD]; V stored TRANSPOSED f16 [B,NH,HD,S] so the
// attention kernel can load PV B-fragments as contiguous 32B vectors.
// ---------------------------------------------------------------------------
__global__ __launch_bounds__(128) void mha_qkv_gemm(
    const float* __restrict__ X, const float* __restrict__ W,
    const float* __restrict__ bias,
    f16* __restrict__ Qo, f16* __restrict__ Ko, f16* __restrict__ Vt)
{
    constexpr int LDA  = 40;  // halves per A-tile row (64x32), 16B-aligned rows
    constexpr int LDBT = 48;  // halves per transposed-B row (64 cols x 32 k), 32B-aligned
    __shared__ f16 sA[64 * LDA];
    __shared__ f16 sBt[64 * LDBT];

    const int tid  = threadIdx.x;
    const int wave = tid >> 5;
    const int lane = tid & 31;
    const int l    = lane & 15;
    const int hi   = lane >> 4;

    const int m0 = blockIdx.x * 64;   // row tile over B*S = 4096
    const int n0 = blockIdx.y * 64;   // col tile over 3*HID = 3072

    v8f acc[4] = {};

    const int arow = tid >> 1, acb = (tid & 1) * 16;  // A loader coords
    const int brow = tid >> 2, bcb = (tid & 3) * 16;  // B loader coords

    for (int kk = 0; kk < HID / 32; ++kk) {
        const int k0 = kk * 32;
        const float* ga = X + (size_t)(m0 + arow) * HID + k0 + acb;
        const float* gb = W + (size_t)(k0 + brow) * (3 * HID) + n0 + bcb;

        __syncthreads();
        f16* la = sA + arow * LDA + acb;
#pragma unroll
        for (int j = 0; j < 16; ++j) la[j] = (f16)ga[j];
#pragma unroll
        for (int j = 0; j < 16; ++j)            // transpose into [n][k]
            sBt[(bcb + j) * LDBT + brow] = (f16)gb[j];
        if (kk + 1 < HID / 32) {                // next-tile prefetch
            __builtin_prefetch(ga + 32, 0, 1);
            __builtin_prefetch(gb + (size_t)32 * (3 * HID), 0, 1);
        }
        __syncthreads();

        // A fragment: 16x32 f16, rows = wave strip
        const f16* Ar = sA + (wave * 16 + l) * LDA;
        const int klo = hi * 16;
        v16h a;
#pragma unroll
        for (int j = 0; j < 8; ++j) { a[j] = Ar[klo + j]; a[8 + j] = Ar[8 + klo + j]; }

#pragma unroll
        for (int nt = 0; nt < 4; ++nt) {
            // B fragment: lane = column, 16 contiguous k-halves (32B aligned)
            const v16h b = *(const v16h*)(sBt + (nt * 16 + l) * LDBT + hi * 16);
            acc[nt] = WMMA_F32_F16(a, b, acc[nt]);
        }
    }

    // Epilogue: add bias, scatter into Q/K [B,NH,S,HD] and V^T [B,NH,HD,S]
    const int rowbase = m0 + wave * 16 + hi * 8;
#pragma unroll
    for (int nt = 0; nt < 4; ++nt) {
        const int col   = n0 + nt * 16 + l;
        const int which = col >> 10;          // 0=Q 1=K 2=V
        const int hc    = col & 1023;
        const int head  = hc >> 6;
        const int dd    = hc & 63;
        const float bv = bias[col];
#pragma unroll
        for (int r = 0; r < 8; ++r) {
            const int row = rowbase + r;
            const int bb = row >> 11;         // / Sseq
            const int ss = row & 2047;
            const f16 val = (f16)(acc[nt][r] + bv);
            if (which == 2) {
                Vt[(((size_t)bb * NH + head) * HD + dd) * Sseq + ss] = val;
            } else {
                f16* dst = (which == 0) ? Qo : Ko;
                dst[(((size_t)bb * NH + head) * Sseq + ss) * HD + dd] = val;
            }
        }
    }
}

// ---------------------------------------------------------------------------
// Kernel 2: causal flash attention per (b, h, 64-query block).
// 4 waves, each owns a 16-row Q strip.  attn out as f16 [B*S, HID].
// ---------------------------------------------------------------------------
__global__ __launch_bounds__(128) void mha_flash_attn(
    const f16* __restrict__ Qf, const f16* __restrict__ Kf,
    const f16* __restrict__ Vt, f16* __restrict__ attn)
{
    constexpr int PS = 72;                 // padded stride of P staging rows
    __shared__ f16 sP[4 * 16 * PS];

    const int tid  = threadIdx.x;
    const int wave = tid >> 5;
    const int lane = tid & 31;
    const int l    = lane & 15;
    const int hi   = lane >> 4;

    const int qb = blockIdx.x;             // 64-row query tile (0..31)
    const int h  = blockIdx.y;
    const int bz = blockIdx.z;
    const size_t base = ((size_t)bz * NH + h) * Sseq * HD;  // same for [S,D] and [D,S]
    const int q0   = qb * 64;
    const int qrow = q0 + wave * 16;
    const float NEG_INF = -__builtin_inff();

    // Q strip as two 16x32 A-fragments (d-chunks)
    const f16* qp = Qf + base + (size_t)(qrow + l) * HD;
    const int klo = hi * 16;
    v16h aQ0, aQ1;
#pragma unroll
    for (int j = 0; j < 8; ++j) {
        aQ0[j] = qp[klo + j];       aQ0[8 + j] = qp[8 + klo + j];
        aQ1[j] = qp[32 + klo + j];  aQ1[8 + j] = qp[40 + klo + j];
    }

    float mrow[8], lrow[8];
    v8f o[4] = {};
#pragma unroll
    for (int r = 0; r < 8; ++r) { mrow[r] = NEG_INF; lrow[r] = 0.0f; }

    f16* Pw = sP + wave * 16 * PS;

    for (int kt = 0; kt <= qb; ++kt) {     // causal: only tiles up to diagonal
        // ---- scores: Q @ K^T  (K rows are contiguous d-slices -> B frags) --
        v8f zv[4];
#pragma unroll
        for (int nt = 0; nt < 4; ++nt) {
            const f16* kp = Kf + base + (size_t)(kt * 64 + nt * 16 + l) * HD;
            const v16h b0 = *(const v16h*)(kp + hi * 16);
            const v16h b1 = *(const v16h*)(kp + 32 + hi * 16);
            v8f s = {};
            s = WMMA_F32_F16(aQ0, b0, s);
            s = WMMA_F32_F16(aQ1, b1, s);
            zv[nt] = s;
        }

        const bool diag = (kt == qb);
        float alpha[8];
#pragma unroll
        for (int r = 0; r < 8; ++r) {
            const int rowg = qrow + hi * 8 + r;
            float rm = NEG_INF;
#pragma unroll
            for (int nt = 0; nt < 4; ++nt) {
                const int key = kt * 64 + nt * 16 + l;
                float z = zv[nt][r] * SCALE;
                z = (diag && key > rowg) ? NEG_INF : z;
                zv[nt][r] = z;
                rm = fmaxf(rm, z);
            }
#pragma unroll
            for (int off = 8; off >= 1; off >>= 1)     // half-wave row max
                rm = fmaxf(rm, __shfl_xor(rm, off, 32));
            const float mn = fmaxf(mrow[r], rm);
            alpha[r] = __expf(mrow[r] - mn);
            mrow[r] = mn;
        }

        // ---- P = exp(z - m); stage to LDS for A-layout reload -------------
#pragma unroll
        for (int r = 0; r < 8; ++r) {
            float rs = 0.0f;
#pragma unroll
            for (int nt = 0; nt < 4; ++nt) {
                const float p = __expf(zv[nt][r] - mrow[r]);
                rs += p;
                Pw[(hi * 8 + r) * PS + nt * 16 + l] = (f16)p;
            }
#pragma unroll
            for (int off = 8; off >= 1; off >>= 1)
                rs += __shfl_xor(rs, off, 32);
            lrow[r] = lrow[r] * alpha[r] + rs;
#pragma unroll
            for (int nt = 0; nt < 4; ++nt) o[nt][r] *= alpha[r];
        }

        asm volatile("s_wait_dscnt 0" ::: "memory");   // intra-wave LDS RAW

        v16h aP0, aP1;
        const f16* Pr = Pw + l * PS;
#pragma unroll
        for (int j = 0; j < 8; ++j) {
            aP0[j] = Pr[klo + j];       aP0[8 + j] = Pr[8 + klo + j];
            aP1[j] = Pr[32 + klo + j];  aP1[8 + j] = Pr[40 + klo + j];
        }

        // ---- O += P @ V  (V^T rows are contiguous key-slices -> B frags) --
#pragma unroll
        for (int nt = 0; nt < 4; ++nt) {
            const f16* vp = Vt + base + (size_t)(nt * 16 + l) * Sseq + kt * 64 + hi * 16;
            const v16h bv0 = *(const v16h*)(vp);       // keys 0..31 half
            const v16h bv1 = *(const v16h*)(vp + 32);  // keys 32..63 half
            o[nt] = WMMA_F32_F16(aP0, bv0, o[nt]);
            o[nt] = WMMA_F32_F16(aP1, bv1, o[nt]);
        }
    }

    // normalize and write attn as f16 [B*S, HID] (row-major for out GEMM)
#pragma unroll
    for (int nt = 0; nt < 4; ++nt) {
#pragma unroll
        for (int r = 0; r < 8; ++r) {
            const int rowg = qrow + hi * 8 + r;
            const float val = o[nt][r] / lrow[r];
            attn[((size_t)bz * Sseq + rowg) * HID + h * HD + nt * 16 + l] = (f16)val;
        }
    }
}

// ---------------------------------------------------------------------------
// Kernel 3: output projection.  attn[4096,1024](f16) @ W_out[1024,1024] + b.
// A tile (already f16) is staged via async global->LDS DMA (ASYNCcnt path);
// B tile is converted f32->f16 and transposed through registers.
// ---------------------------------------------------------------------------
__global__ __launch_bounds__(128) void mha_out_gemm(
    const f16* __restrict__ A, const float* __restrict__ W,
    const float* __restrict__ bias, float* __restrict__ Out)
{
    constexpr int LDAu = 32;  // unpadded: async copy preserves row layout (64B rows)
    constexpr int LDBT = 48;
    __shared__ f16 sA[64 * LDAu];
    __shared__ f16 sBt[64 * LDBT];

    const int tid  = threadIdx.x;
    const int wave = tid >> 5;
    const int lane = tid & 31;
    const int l    = lane & 15;
    const int hi   = lane >> 4;

    const int m0 = blockIdx.x * 64;
    const int n0 = blockIdx.y * 64;

    v8f acc[4] = {};

    const int brow = tid >> 2, bcb = (tid & 3) * 16;

    for (int kk = 0; kk < HID / 32; ++kk) {
        const int k0 = kk * 32;
        const float* gb = W + (size_t)(k0 + brow) * HID + n0 + bcb;

        __syncthreads();

        // ---- A tile: 64x32 f16 (4KB) via 2 async 16B DMA ops per thread ----
#pragma unroll
        for (int i = 0; i < 2; ++i) {
            const int idx = tid * 2 + i;            // 256 x 16B chunks
            const int row = idx >> 2;
            const int seg = idx & 3;
            const f16* gsrc = A + (size_t)(m0 + row) * HID + k0 + seg * 8;
            const unsigned lds_off =
                (unsigned)(size_t)(&sA[row * LDAu + seg * 8]);
            const unsigned long long ga64 = (unsigned long long)(size_t)gsrc;
            asm volatile("global_load_async_to_lds_b128 %0, %1, off"
                         :: "v"(lds_off), "v"(ga64) : "memory");
        }

        // ---- B tile: f32 -> f16, transposed into [n][k] ---------------------
#pragma unroll
        for (int j = 0; j < 16; ++j)
            sBt[(bcb + j) * LDBT + brow] = (f16)gb[j];
        if (kk + 1 < HID / 32)
            __builtin_prefetch(gb + (size_t)32 * HID, 0, 1);

        asm volatile("s_wait_asynccnt 0" ::: "memory");  // A DMA landed in LDS
        __syncthreads();

        const f16* Ar = sA + (wave * 16 + l) * LDAu;
        const int klo = hi * 16;
        v16h a;
#pragma unroll
        for (int j = 0; j < 8; ++j) { a[j] = Ar[klo + j]; a[8 + j] = Ar[8 + klo + j]; }

#pragma unroll
        for (int nt = 0; nt < 4; ++nt) {
            const v16h b = *(const v16h*)(sBt + (nt * 16 + l) * LDBT + hi * 16);
            acc[nt] = WMMA_F32_F16(a, b, acc[nt]);
        }
    }

    const int rowbase = m0 + wave * 16 + hi * 8;
#pragma unroll
    for (int nt = 0; nt < 4; ++nt) {
        const int col = n0 + nt * 16 + l;
        const float bv = bias[col];
#pragma unroll
        for (int r = 0; r < 8; ++r)
            Out[(size_t)(rowbase + r) * HID + col] = acc[nt][r] + bv;
    }
}

// ---------------------------------------------------------------------------
extern "C" void kernel_launch(void* const* d_in, const int* in_sizes, int n_in,
                              void* d_out, int out_size, void* d_ws, size_t ws_size,
                              hipStream_t stream) {
    const float* x     = (const float*)d_in[0];
    const float* W_qkv = (const float*)d_in[1];
    const float* b_qkv = (const float*)d_in[2];
    const float* W_out = (const float*)d_in[3];
    const float* b_out = (const float*)d_in[4];
    float* out = (float*)d_out;

    const size_t TEN = (size_t)Bsz * NH * Sseq * HD;   // 4,194,304 halves each
    f16* Q    = (f16*)d_ws;
    f16* K    = Q + TEN;
    f16* V    = K + TEN;                               // V stored transposed [B,NH,HD,S]
    f16* attn = V + TEN;                               // 32 MB total workspace

    // 1) QKV projection: [4096,1024] x [1024,3072]
    mha_qkv_gemm<<<dim3((Bsz * Sseq) / 64, (3 * HID) / 64), 128, 0, stream>>>(
        x, W_qkv, b_qkv, Q, K, V);

    // 2) causal flash attention per (q-tile, head, batch)
    mha_flash_attn<<<dim3(Sseq / 64, NH, Bsz), 128, 0, stream>>>(Q, K, V, attn);

    // 3) output projection: [4096,1024] x [1024,1024]
    mha_out_gemm<<<dim3((Bsz * Sseq) / 64, HID / 64), 128, 0, stream>>>(
        attn, W_out, b_out, out);
}
